// EfficientTransformerQNetwork_18330920419661
// MI455X (gfx1250) — matrix operand order repaired and checked by reading, not compiled, not run on hardware
//
#include <hip/hip_runtime.h>
#include <hip/hip_bf16.h>

// ---------------------------------------------------------------------------
// Performer (FAVOR+) forward for gfx1250 using v_wmma_f32_16x16x32_bf16.
// B=4, N=8192, DIM=512, HEADS=8, DH=64, NB=266 (padded to 272), FF=2048, OUT=32
// ---------------------------------------------------------------------------

typedef __bf16 bf16;
typedef __attribute__((ext_vector_type(8)))  bf16  bf16x8;
typedef __attribute__((ext_vector_type(16))) bf16  bf16x16;
typedef __attribute__((ext_vector_type(8)))  float f32x8;
typedef __attribute__((ext_vector_type(4)))  int   v4i;

union V16 { bf16x16 v; bf16x8 h[2]; };

#define F_BIAS   1
#define F_RES    2
#define F_GELU   4
#define F_ROWSC  8
#define F_OUTB   16
#define F_OUTF   32
#define F_ATOMIC 64

// CDNA5 async global->LDS copies (ASYNCcnt-tracked); guarded so the file
// still compiles if this toolchain doesn't declare the builtins.
// Probe result (round 2 diagnostic): param 0 is 'int4 __device__*' (AS1),
// so pass typed v4i pointers in address spaces 1 / 3.
#if defined(__has_builtin)
#if __has_builtin(__builtin_amdgcn_global_load_async_to_lds_b128) && \
    __has_builtin(__builtin_amdgcn_s_wait_asynccnt)
#define USE_ASYNC_LDS 1
#endif
#endif
#ifndef USE_ASYNC_LDS
#define USE_ASYNC_LDS 0
#endif

typedef __attribute__((address_space(1))) v4i gas_v4i;   // global AS
typedef __attribute__((address_space(3))) v4i las_v4i;   // LDS AS

__device__ __forceinline__ float gelu_f(float x) {
    float x3 = x * x * x;
    return 0.5f * x * (1.f + tanhf(0.7978845608028654f * (x + 0.044715f * x3)));
}

// ---------------------------------------------------------------------------
// Generic batched WMMA GEMM: C[z] = A[z] * B[z] (+bias, +residual, gelu, ...)
// A: bf16 row-major [M,K] (or [K,M] if TRANSA), B: bf16 row-major [K,N].
// Block tile 128x128x32, 8 waves, each wave 64x32 (4x2 WMMA accumulators).
// grid.z = batch * ksplit; batch decomposed as (b,h) via Hdim with per-operand
// strides so per-head views of [n, h*64+d] tensors need no repacking.
// Interior tiles take an unguarded fast path (async LDS copy for A + prefetch);
// edge tiles take the fully guarded path.
// ---------------------------------------------------------------------------
template<int FLAGS, bool TRANSA>
__global__ __launch_bounds__(256) void gemm_wmma(
    const bf16* __restrict__ A, const bf16* __restrict__ B,
    const float* __restrict__ bias, const float* __restrict__ resid,
    const float* __restrict__ rowsc,
    float* __restrict__ outF, bf16* __restrict__ outB,
    int M, int N, int K, int lda, int ldb, int ldc,
    int Hdim, long sAb, long sAh, long sBb, long sBh, long sCb, long sCh,
    long sRb, long sRh, int ksplit, int kchunk)
{
    __shared__ __attribute__((aligned(16))) bf16 As[128 * 40];
    __shared__ __attribute__((aligned(16))) bf16 Bs[128 * 40];

    const int tid   = threadIdx.x;
    const int z     = blockIdx.z;
    const int batch = z / ksplit, kseg = z % ksplit;
    const int bb    = batch / Hdim, hh = batch % Hdim;
    A += (long)bb * sAb + (long)hh * sAh;
    B += (long)bb * sBb + (long)hh * sBh;
    const long coff = (long)bb * sCb + (long)hh * sCh;
    const float* rs = (FLAGS & F_ROWSC) ? rowsc + (long)bb * sRb + (long)hh * sRh : nullptr;

    const int kbeg = kseg * kchunk;
    int kend = kbeg + kchunk; if (kend > K) kend = K;

    const int m0 = blockIdx.y * 128, n0 = blockIdx.x * 128;
    const int wid = tid >> 5, lane = tid & 31;
    const int wm = wid & 1, wn = wid >> 1;          // waves: 2 x 4
    const int lm = lane & 15, half = lane >> 4;
    const bool mnfull = (m0 + 128 <= M) && (n0 + 128 <= N);

    f32x8 acc[4][2] = {};

    for (int kt = kbeg; kt < kend; kt += 32) {
        if (mnfull && (kt + 32 <= kend)) {
            // ---------------- interior fast path ----------------
            const bool more = (kt + 32 < kend);
            #pragma unroll
            for (int i = 0; i < 2; ++i) {
                int c = tid * 2 + i;                // 512 chunks of 8 halves
                if (!TRANSA) {
                    int row = c >> 2, kc = (c & 3) << 3;
                    const bf16* gp = A + (long)(m0 + row) * lda + (kt + kc);
#if USE_ASYNC_LDS
                    __builtin_amdgcn_global_load_async_to_lds_b128(
                        (gas_v4i*)(bf16*)gp,
                        (las_v4i*)&As[row * 40 + kc],
                        0, 0);
#else
                    *(bf16x8*)(&As[row * 40 + kc]) = *(const bf16x8*)gp;
#endif
                    if (more) __builtin_prefetch(gp + 32, 0, 3);
                } else {
                    int krow = c >> 4, mc = (c & 15) << 3;
                    const bf16* gp = A + (long)(kt + krow) * lda + (m0 + mc);
                    bf16x8 v = *(const bf16x8*)gp;
                    #pragma unroll
                    for (int j = 0; j < 8; ++j) As[(mc + j) * 40 + krow] = v[j];
                    if (more) __builtin_prefetch(gp + (long)32 * lda, 0, 3);
                }
                {
                    int krow = c >> 4, nc = (c & 15) << 3;
                    const bf16* gp = B + (long)(kt + krow) * ldb + (n0 + nc);
                    bf16x8 v = *(const bf16x8*)gp;
                    #pragma unroll
                    for (int j = 0; j < 8; ++j) Bs[(nc + j) * 40 + krow] = v[j];
                    if (more) __builtin_prefetch(gp + (long)32 * ldb, 0, 3);
                }
            }
#if USE_ASYNC_LDS
            if (!TRANSA) __builtin_amdgcn_s_wait_asynccnt(0);
#endif
        } else {
            // ---------------- guarded edge path ----------------
            #pragma unroll
            for (int i = 0; i < 2; ++i) {
                int c = tid * 2 + i;
                if (!TRANSA) {
                    int row = c >> 2, kc = (c & 3) << 3;
                    int gm = m0 + row, gk = kt + kc;
                    bf16x8 v = {};
                    if (gm < M) {
                        if (gk + 8 <= kend) {
                            v = *(const bf16x8*)(A + (long)gm * lda + gk);
                        } else {
                            for (int j = 0; j < 8; ++j)
                                if (gk + j < kend) v[j] = A[(long)gm * lda + gk + j];
                        }
                    }
                    *(bf16x8*)(&As[row * 40 + kc]) = v;
                } else {
                    int krow = c >> 4, mc = (c & 15) << 3;
                    int gk = kt + krow, gm = m0 + mc;
                    bf16x8 v = {};
                    if (gk < kend) {
                        if (gm + 8 <= M) {
                            v = *(const bf16x8*)(A + (long)gk * lda + gm);
                        } else {
                            for (int j = 0; j < 8; ++j)
                                if (gm + j < M) v[j] = A[(long)gk * lda + gm + j];
                        }
                    }
                    for (int j = 0; j < 8; ++j) As[(mc + j) * 40 + krow] = v[j];
                }
                {
                    int krow = c >> 4, nc = (c & 15) << 3;
                    int gk = kt + krow, gn = n0 + nc;
                    bf16x8 v = {};
                    if (gk < kend) {
                        if (gn + 8 <= N) {
                            v = *(const bf16x8*)(B + (long)gk * ldb + gn);
                        } else {
                            for (int j = 0; j < 8; ++j)
                                if (gn + j < N) v[j] = B[(long)gk * ldb + gn + j];
                        }
                    }
                    for (int j = 0; j < 8; ++j) Bs[(nc + j) * 40 + krow] = v[j];
                }
            }
        }
        __syncthreads();

        // ---- load fragments (A 16x32, B 32x16 bf16 lane layouts) ----
        V16 af[4], bfr[2];
        #pragma unroll
        for (int i = 0; i < 4; ++i) {
            int m = wm * 64 + i * 16 + lm;
            af[i].h[0] = *(const bf16x8*)(&As[m * 40 + half * 8]);
            af[i].h[1] = *(const bf16x8*)(&As[m * 40 + half * 8 + 16]);
        }
        #pragma unroll
        for (int j = 0; j < 2; ++j) {
            int n = wn * 32 + j * 16 + lm;
            bfr[j].h[0] = *(const bf16x8*)(&Bs[n * 40 + half * 8]);
            bfr[j].h[1] = *(const bf16x8*)(&Bs[n * 40 + half * 8 + 16]);
        }
        #pragma unroll
        for (int i = 0; i < 4; ++i)
            #pragma unroll
            for (int j = 0; j < 2; ++j)
                acc[i][j] = __builtin_amdgcn_wmma_f32_16x16x32_bf16(
                    false, af[i].v, false, bfr[j].v, (short)0, acc[i][j], false, false);
        __syncthreads();
    }

    // ---- epilogue: C layout lane = n%16, half selects m+8, VGPR r = m ----
    #pragma unroll
    for (int i = 0; i < 4; ++i) {
        int gmb = m0 + wm * 64 + i * 16 + half * 8;
        #pragma unroll
        for (int j = 0; j < 2; ++j) {
            int gn = n0 + wn * 32 + j * 16 + lm;
            if (gn >= N) continue;
            float bval = (FLAGS & F_BIAS) ? bias[gn] : 0.f;
            #pragma unroll
            for (int r = 0; r < 8; ++r) {
                int gm = gmb + r;
                if (gm >= M) continue;
                float v = acc[i][j][r] + bval;
                if (FLAGS & F_GELU)  v = gelu_f(v);
                if (FLAGS & F_ROWSC) v *= rs[gm];
                long idx = coff + (long)gm * ldc + gn;
                if (FLAGS & F_RES)   v += resid[idx];
                if (FLAGS & F_ATOMIC)      atomicAdd(outF + idx, v);
                else if (FLAGS & F_OUTF)   outF[idx] = v;
                if (FLAGS & F_OUTB)        outB[idx] = (bf16)v;
            }
        }
    }
}

// ---------------------------------------------------------------------------
// LayerNorm over rows of 512, one wave per row, bf16 output.
// ---------------------------------------------------------------------------
__global__ __launch_bounds__(256) void ln_kernel(
    const float* __restrict__ x, const float* __restrict__ g,
    const float* __restrict__ bta, bf16* __restrict__ out, int rows)
{
    int wid = threadIdx.x >> 5, lane = threadIdx.x & 31;
    int row = blockIdx.x * 8 + wid;
    if (row >= rows) return;
    const float* xr = x + (long)row * 512;
    float v[16], s = 0.f, ss = 0.f;
    #pragma unroll
    for (int i = 0; i < 16; ++i) {
        v[i] = xr[lane + i * 32]; s += v[i]; ss += v[i] * v[i];
    }
    #pragma unroll
    for (int o = 16; o > 0; o >>= 1) { s += __shfl_xor(s, o, 32); ss += __shfl_xor(ss, o, 32); }
    float mu   = s * (1.f / 512.f);
    float var  = ss * (1.f / 512.f) - mu * mu;
    float rstd = rsqrtf(var + 1e-5f);
    bf16* orow = out + (long)row * 512;
    #pragma unroll
    for (int i = 0; i < 16; ++i) {
        int e = lane + i * 32;
        orow[e] = (bf16)((v[i] - mu) * rstd * g[e] + bta[e]);
    }
}

// ---------------------------------------------------------------------------
// FAVOR+ exp kernels. xp rows are [.., 272] f32 (cols >= 266 ignored).
// QUERY: per-row max; KEY: global max via order-preserving uint encoding.
// ---------------------------------------------------------------------------
template<bool QUERY>
__global__ __launch_bounds__(256) void favor_exp_kernel(
    const float* __restrict__ xp, const bf16* __restrict__ xsrc,
    const unsigned* __restrict__ gmax, bf16* __restrict__ out)
{
    const float DN2   = 0.125f;               // DH^-0.5
    const float RATIO = 0.06131393394849658f; // NB^-0.5
    int wid = threadIdx.x >> 5, lane = threadIdx.x & 31;
    long row = (long)blockIdx.x * 8 + wid;
    int z = (int)(row >> 13), n = (int)(row & 8191);
    int b = z >> 3, h = z & 7;
    const bf16* xr = xsrc + ((long)(b * 8192 + n)) * 512 + h * 64;
    float x0 = (float)xr[lane], x1 = (float)xr[lane + 32];
    float ss = x0 * x0 + x1 * x1;
    #pragma unroll
    for (int o = 16; o > 0; o >>= 1) ss += __shfl_xor(ss, o, 32);
    float diag = 0.5f * ss * DN2;

    const float* xprow = xp + row * 272;
    float xv[9], mx = -3.0e38f;
    #pragma unroll
    for (int it = 0; it < 9; ++it) {
        int m = lane + it * 32;
        xv[it] = (m < 272) ? xprow[m] : -3.0e38f;
        if (m < 266) mx = fmaxf(mx, xv[it]);
    }
    if (QUERY) {
        #pragma unroll
        for (int o = 16; o > 0; o >>= 1) mx = fmaxf(mx, __shfl_xor(mx, o, 32));
    } else {
        unsigned u = *gmax;
        unsigned bits = (u & 0x80000000u) ? (u ^ 0x80000000u) : ~u;
        mx = __uint_as_float(bits);
    }
    bf16* orow = out + row * 272;
    #pragma unroll
    for (int it = 0; it < 9; ++it) {
        int m = lane + it * 32;
        if (m < 272) {
            float r = (m < 266) ? RATIO * (expf(xv[it] - diag - mx) + 1e-4f) : 0.f;
            orow[m] = (bf16)r;
        }
    }
}

__global__ __launch_bounds__(256) void favor_max_kernel(
    const float* __restrict__ xp, unsigned* gmax)
{
    int wid = threadIdx.x >> 5, lane = threadIdx.x & 31;
    long row = (long)blockIdx.x * 8 + wid;
    const float* xprow = xp + row * 272;
    float mx = -3.0e38f;
    for (int m = lane; m < 266; m += 32) mx = fmaxf(mx, xprow[m]);
    #pragma unroll
    for (int o = 16; o > 0; o >>= 1) mx = fmaxf(mx, __shfl_xor(mx, o, 32));
    if (lane == 0) {
        unsigned bits = __float_as_uint(mx);
        unsigned u = (bits & 0x80000000u) ? ~bits : (bits | 0x80000000u);
        atomicMax(gmax, u);
    }
}

__global__ void init_max_kernel(unsigned* g) { *g = 0u; }

// k_sum[z][m] = sum_n kp[z][n][m]
__global__ __launch_bounds__(256) void ksum_kernel(
    const bf16* __restrict__ kp, float* __restrict__ ks)
{
    int z = blockIdx.x;
    for (int m = threadIdx.x; m < 272; m += 256) {
        const bf16* base = kp + (long)z * 8192 * 272 + m;
        float s = 0.f;
        for (int n = 0; n < 8192; ++n) s += (float)base[(long)n * 272];
        ks[z * 272 + m] = s;
    }
}

// dinv[row] = 1 / dot(qp[row], ksum[z])
__global__ __launch_bounds__(256) void dinv_kernel(
    const bf16* __restrict__ qp, const float* __restrict__ ks,
    float* __restrict__ dinv)
{
    int wid = threadIdx.x >> 5, lane = threadIdx.x & 31;
    long row = (long)blockIdx.x * 8 + wid;
    int z = (int)(row >> 13);
    const bf16* qr = qp + row * 272;
    const float* kr = ks + z * 272;
    float s = 0.f;
    for (int m = lane; m < 272; m += 32) s += (float)qr[m] * kr[m];
    #pragma unroll
    for (int o = 16; o > 0; o >>= 1) s += __shfl_xor(s, o, 32);
    if (lane == 0) dinv[row] = 1.f / s;
}

// ---------------------------------------------------------------------------
// Utility kernels
// ---------------------------------------------------------------------------
__global__ void copy_f32_kernel(const float* __restrict__ a, float* __restrict__ b, long n) {
    long i = (long)blockIdx.x * 256 + threadIdx.x; if (i < n) b[i] = a[i];
}
__global__ void cvt_bf16_kernel(const float* __restrict__ a, bf16* __restrict__ b, long n) {
    long i = (long)blockIdx.x * 256 + threadIdx.x; if (i < n) b[i] = (bf16)a[i];
}
__global__ void zero_f32_kernel(float* __restrict__ a, long n) {
    long i = (long)blockIdx.x * 256 + threadIdx.x; if (i < n) a[i] = 0.f;
}
// projT[l][d][m] = proj[l][m][d] * DH^-0.25, zero-padded m in [266,272)
__global__ void build_projT_kernel(const float* __restrict__ proj, bf16* __restrict__ pT) {
    long i = (long)blockIdx.x * 256 + threadIdx.x;
    const long total = 6l * 64 * 272;
    if (i >= total) return;
    int m = (int)(i % 272); long t = i / 272;
    int d = (int)(t % 64);  int  l = (int)(t / 64);
    float v = 0.f;
    if (m < 266) v = proj[((long)l * 266 + m) * 64 + d] * 0.35355339059327373f;
    pT[i] = (bf16)v;
}

// ---------------------------------------------------------------------------
extern "C" void kernel_launch(void* const* d_in, const int* in_sizes, int n_in,
                              void* d_out, int out_size, void* d_ws, size_t ws_size,
                              hipStream_t stream)
{
    (void)in_sizes; (void)n_in; (void)out_size; (void)ws_size;
    const float* src  = (const float*)d_in[0];
    const float* proj = (const float*)d_in[1];
    const float* ln1g = (const float*)d_in[2];
    const float* ln1b = (const float*)d_in[3];
    const float* Wq   = (const float*)d_in[4];
    const float* Wk   = (const float*)d_in[5];
    const float* Wv   = (const float*)d_in[6];
    const float* Wo   = (const float*)d_in[7];
    const float* bo   = (const float*)d_in[8];
    const float* ln2g = (const float*)d_in[9];
    const float* ln2b = (const float*)d_in[10];
    const float* W1   = (const float*)d_in[11];
    const float* b1   = (const float*)d_in[12];
    const float* W2   = (const float*)d_in[13];
    const float* b2   = (const float*)d_in[14];
    const float* fcw  = (const float*)d_in[15];
    const float* fcb  = (const float*)d_in[16];
    float* outp = (float*)d_out;

    const long BN = 32768;   // B*N tokens
    size_t off = 0;
    auto carve = [&](size_t bytes) -> char* {
        char* p = (char*)d_ws + off; off += (bytes + 255) & ~(size_t)255; return p;
    };
    float* xbuf  = (float*)carve(BN * 512 * 4);
    bf16*  hbuf  = (bf16*) carve(BN * 512 * 2);
    bf16*  qbuf  = (bf16*) carve(BN * 512 * 2);
    bf16*  kbuf  = (bf16*) carve(BN * 512 * 2);
    bf16*  vbuf  = (bf16*) carve(BN * 512 * 2);
    bf16*  obuf  = (bf16*) carve(BN * 512 * 2);
    bf16*  ffbuf = (bf16*) carve(BN * 2048 * 2);
    float* xpbuf = (float*)carve(32l * 8192 * 272 * 4);
    bf16*  qpbuf = (bf16*) carve(32l * 8192 * 272 * 2);
    bf16*  kpbuf = (bf16*) carve(32l * 8192 * 272 * 2);
    float* ctxf  = (float*)carve(32l * 272 * 64 * 4);
    bf16*  ctxb  = (bf16*) carve(32l * 272 * 64 * 2);
    float* ksumb = (float*)carve(32l * 272 * 4);
    float* dinvb = (float*)carve(32l * 8192 * 4);
    unsigned* maxb = (unsigned*)carve(256);
    bf16* wq_bf = (bf16*)carve(6l * 512 * 512 * 2);
    bf16* wk_bf = (bf16*)carve(6l * 512 * 512 * 2);
    bf16* wv_bf = (bf16*)carve(6l * 512 * 512 * 2);
    bf16* wo_bf = (bf16*)carve(6l * 512 * 512 * 2);
    bf16* w1_bf = (bf16*)carve(6l * 512 * 2048 * 2);
    bf16* w2_bf = (bf16*)carve(6l * 512 * 2048 * 2);
    bf16* fc_bf = (bf16*)carve(512l * 32 * 2);
    bf16* pT    = (bf16*)carve(6l * 64 * 272 * 2);

    auto blk = [](long n) { return (unsigned)((n + 255) / 256); };

    // weight conversion (every call; deterministic)
    long nw = 6l * 512 * 512, nf = 6l * 512 * 2048;
    cvt_bf16_kernel<<<blk(nw), 256, 0, stream>>>(Wq, wq_bf, nw);
    cvt_bf16_kernel<<<blk(nw), 256, 0, stream>>>(Wk, wk_bf, nw);
    cvt_bf16_kernel<<<blk(nw), 256, 0, stream>>>(Wv, wv_bf, nw);
    cvt_bf16_kernel<<<blk(nw), 256, 0, stream>>>(Wo, wo_bf, nw);
    cvt_bf16_kernel<<<blk(nf), 256, 0, stream>>>(W1, w1_bf, nf);
    cvt_bf16_kernel<<<blk(nf), 256, 0, stream>>>(W2, w2_bf, nf);
    cvt_bf16_kernel<<<blk(512 * 32), 256, 0, stream>>>(fcw, fc_bf, 512 * 32);
    build_projT_kernel<<<blk(6l * 64 * 272), 256, 0, stream>>>(proj, pT);
    copy_f32_kernel<<<blk(BN * 512), 256, 0, stream>>>(src, xbuf, BN * 512);

    const long SQ = 8192l * 512;   // per-batch token stride in [BN,512] tensors
    const long SP = 8192l * 272;   // per-head stride in [z,8192,272] tensors
    const long SC = 272l * 64;     // per-head stride in ctx

    for (int l = 0; l < 6; ++l) {
        const bf16* wq = wq_bf + (long)l * 512 * 512;
        const bf16* wk = wk_bf + (long)l * 512 * 512;
        const bf16* wv = wv_bf + (long)l * 512 * 512;
        const bf16* wo = wo_bf + (long)l * 512 * 512;
        const bf16* w1 = w1_bf + (long)l * 512 * 2048;
        const bf16* w2 = w2_bf + (long)l * 2048 * 512;
        const bf16* pt = pT + (long)l * 64 * 272;

        // ---- LN1 -> h ----
        ln_kernel<<<4096, 256, 0, stream>>>(xbuf, ln1g + l * 512, ln1b + l * 512, hbuf, 32768);

        // ---- q/k/v = h * W ----
        dim3 gqkv(4, 256, 1);
        gemm_wmma<F_OUTB, false><<<gqkv, 256, 0, stream>>>(hbuf, wq, nullptr, nullptr, nullptr,
            nullptr, qbuf, 32768, 512, 512, 512, 512, 512, 1, 0, 0, 0, 0, 0, 0, 0, 0, 1, 512);
        gemm_wmma<F_OUTB, false><<<gqkv, 256, 0, stream>>>(hbuf, wk, nullptr, nullptr, nullptr,
            nullptr, kbuf, 32768, 512, 512, 512, 512, 512, 1, 0, 0, 0, 0, 0, 0, 0, 0, 1, 512);
        gemm_wmma<F_OUTB, false><<<gqkv, 256, 0, stream>>>(hbuf, wv, nullptr, nullptr, nullptr,
            nullptr, vbuf, 32768, 512, 512, 512, 512, 512, 1, 0, 0, 0, 0, 0, 0, 0, 0, 1, 512);

        // ---- xp_q = q(per-head) * projT; qp = exp(..) ----
        dim3 gxp(3, 64, 32);
        gemm_wmma<F_OUTF, false><<<gxp, 256, 0, stream>>>(qbuf, pt, nullptr, nullptr, nullptr,
            xpbuf, nullptr, 8192, 272, 64, 512, 272, 272,
            8, SQ, 64, 0, 0, 8 * SP, SP, 0, 0, 1, 64);
        favor_exp_kernel<true><<<32768, 256, 0, stream>>>(xpbuf, qbuf, nullptr, qpbuf);

        // ---- xp_k; global max; kp = exp(..) ----
        gemm_wmma<F_OUTF, false><<<gxp, 256, 0, stream>>>(kbuf, pt, nullptr, nullptr, nullptr,
            xpbuf, nullptr, 8192, 272, 64, 512, 272, 272,
            8, SQ, 64, 0, 0, 8 * SP, SP, 0, 0, 1, 64);
        init_max_kernel<<<1, 1, 0, stream>>>(maxb);
        favor_max_kernel<<<32768, 256, 0, stream>>>(xpbuf, maxb);
        favor_exp_kernel<false><<<32768, 256, 0, stream>>>(xpbuf, kbuf, maxb, kpbuf);

        // ---- k_sum, d_inv ----
        ksum_kernel<<<32, 256, 0, stream>>>(kpbuf, ksumb);
        dinv_kernel<<<32768, 256, 0, stream>>>(qpbuf, ksumb, dinvb);

        // ---- ctx = kp^T * v  (split-K=8, f32 atomics) ----
        zero_f32_kernel<<<blk(32l * 272 * 64), 256, 0, stream>>>(ctxf, 32l * 272 * 64);
        dim3 gctx(1, 3, 32 * 8);
        gemm_wmma<F_ATOMIC, true><<<gctx, 256, 0, stream>>>(kpbuf, vbuf, nullptr, nullptr, nullptr,
            ctxf, nullptr, 272, 64, 8192, 272, 512, 64,
            8, 8 * SP, SP, SQ, 64, 8 * SC, SC, 0, 0, 8, 1024);
        cvt_bf16_kernel<<<blk(32l * 272 * 64), 256, 0, stream>>>(ctxf, ctxb, 32l * 272 * 64);

        // ---- o = (qp * ctx) * d_inv, scattered into [n, h*64+d] ----
        dim3 go(1, 64, 32);
        gemm_wmma<F_OUTB | F_ROWSC, false><<<go, 256, 0, stream>>>(qpbuf, ctxb, nullptr, nullptr,
            dinvb, nullptr, obuf, 8192, 64, 272, 272, 64, 512,
            8, 8 * SP, SP, 8 * SC, SC, SQ, 64, 8l * 8192, 8192, 1, 272);

        // ---- x += o * Wo + bo ----
        gemm_wmma<F_OUTF | F_RES | F_BIAS, false><<<gqkv, 256, 0, stream>>>(obuf, wo,
            bo + l * 512, xbuf, nullptr, xbuf, nullptr,
            32768, 512, 512, 512, 512, 512, 1, 0, 0, 0, 0, 0, 0, 0, 0, 1, 512);

        // ---- LN2 -> h; ff = gelu(h*W1+b1); x += ff*W2 + b2 ----
        ln_kernel<<<4096, 256, 0, stream>>>(xbuf, ln2g + l * 512, ln2b + l * 512, hbuf, 32768);
        dim3 gff1(16, 256, 1);
        gemm_wmma<F_OUTB | F_BIAS | F_GELU, false><<<gff1, 256, 0, stream>>>(hbuf, w1,
            b1 + l * 2048, nullptr, nullptr, nullptr, ffbuf,
            32768, 2048, 512, 512, 2048, 2048, 1, 0, 0, 0, 0, 0, 0, 0, 0, 1, 512);
        gemm_wmma<F_OUTF | F_RES | F_BIAS, false><<<gqkv, 256, 0, stream>>>(ffbuf, w2,
            b2 + l * 512, xbuf, nullptr, xbuf, nullptr,
            32768, 512, 2048, 2048, 512, 512, 1, 0, 0, 0, 0, 0, 0, 0, 0, 1, 2048);
    }

    // ---- final fc: out = x * fc_w + fc_b ----
    cvt_bf16_kernel<<<blk(BN * 512), 256, 0, stream>>>(xbuf, hbuf, BN * 512);
    dim3 gfc(1, 256, 1);
    gemm_wmma<F_OUTF | F_BIAS, false><<<gfc, 256, 0, stream>>>(hbuf, fc_bf, fcb,
        nullptr, nullptr, outp, nullptr,
        32768, 32, 512, 512, 32, 32, 1, 0, 0, 0, 0, 0, 0, 0, 0, 1, 512);
}
